// GCN_60842506715219
// MI455X (gfx1250) — matrix operand order repaired and checked by reading, not compiled, not run on hardware
//
#include <hip/hip_runtime.h>
#include <hip/hip_bf16.h>

typedef __bf16 v16bf __attribute__((ext_vector_type(16)));
typedef float  v8f   __attribute__((ext_vector_type(8)));
typedef unsigned int u32x4 __attribute__((ext_vector_type(4)));
typedef int          i32x4 __attribute__((ext_vector_type(4)));
typedef int          i32x8 __attribute__((ext_vector_type(8)));

#define DMKT 64
#define DH   128
#define DG   64
#define NCHUNK 132       // (64*64 + 128) / 32 K-chunks for fused GEMM
#define NCHUNK_GCN 4     // 128 / 32
#define CHUNK_BYTES (DH * 32 * 2)   // 8 KB of bf16 B data per K-chunk

#if __has_builtin(__builtin_amdgcn_tensor_load_to_lds) && __has_builtin(__builtin_amdgcn_s_wait_tensorcnt)
#define HAVE_TDM 1
#else
#define HAVE_TDM 0
#endif

// ---------------------------------------------------------------------------
// Pack T_fuse [128,64,64] (+ W_fuse [128,128] appended as K=4096..4223) into
// bf16 B-fragment order: bp[((c*128 + h)*32) + j'],  j' = half*16 + elem,
// contiguous k order, same as A frag (lane<16: K=c*32+0..15, lane>=16: K=c*32+16..31).
// ---------------------------------------------------------------------------
__global__ void pack_main_kernel(const float* __restrict__ T,
                                 const float* __restrict__ Wf,
                                 __bf16* __restrict__ bp) {
  int idx = blockIdx.x * blockDim.x + threadIdx.x;
  const int total = NCHUNK * DH * 32;
  if (idx >= total) return;
  int jp = idx & 31;
  int h  = (idx >> 5) & (DH - 1);
  int c  = idx >> 12;                 // / (32*128)
  float v;
  if (c < 128) {                      // bilinear part: k = d*64 + e
    int d = c >> 1;
    int e = ((c & 1) << 5) + jp;
    v = T[h * 4096 + d * 64 + e];
  } else {                            // linear part: W_fuse[k', h]
    int k = ((c - 128) << 5) + jp;
    v = Wf[k * DH + h];
  }
  bp[idx] = (__bf16)v;
}

__global__ void pack_gcn_kernel(const float* __restrict__ Wg,
                                __bf16* __restrict__ bp) {
  int idx = blockIdx.x * blockDim.x + threadIdx.x;
  const int total = NCHUNK_GCN * DG * 32;
  if (idx >= total) return;
  int jp = idx & 31;
  int h  = (idx >> 5) & (DG - 1);
  int c  = idx >> 11;                 // / (32*64)
  int k  = c * 32 + jp;
  bp[idx] = (__bf16)Wg[k * DG + h];
}

// agg := 0, deg := 1 (self loops)
__global__ void init_kernel(float* __restrict__ agg, float* __restrict__ deg, int N) {
  int i = blockIdx.x * blockDim.x + threadIdx.x;
  if (i < N * 64) agg[i] = 0.0f;
  if (i < N)      deg[i] = 1.0f;
}

__global__ void degree_kernel(const int* __restrict__ ei, float* __restrict__ deg,
                              int E, int N) {
  int e = blockIdx.x * blockDim.x + threadIdx.x;
  if (e >= E) return;
  int d = ei[E + e];
  if (d >= 0 && d < N) atomicAdd(&deg[d], 1.0f);
}

#if HAVE_TDM
// Issue one TDM DMA: CHUNK_BYTES linear bytes, global -> LDS.
// D# per CDNA5 ISA: group0 = {count, lds_addr, global_addr, type=2},
// group1 = {data_size=4B, tensor_dim0=tile_dim0=2048 dwords, dim1=1, stride}.
__device__ __forceinline__ void tdm_load_chunk(const void* gsrc, unsigned lds_off) {
  unsigned long long ga = (unsigned long long)gsrc;
  const unsigned ndw = CHUNK_BYTES >> 2;                  // 2048 dwords
  u32x4 g0;
  g0.x = 1u;                                              // count = 1 (valid)
  g0.y = lds_off;                                         // lds_addr [63:32]
  g0.z = (unsigned)(ga & 0xffffffffull);                  // global_addr [95:64]
  g0.w = (unsigned)((ga >> 32) & 0x01ffffffull)           // global_addr [120:96]
         | (2u << 30);                                    // type = 2 ("image")
  i32x8 g1;
  g1[0] = (2 << 16);                                      // data_size = 4 bytes
  g1[1] = (int)((ndw & 0xffffu) << 16);                   // tensor_dim0 lo16
  g1[2] = (int)(((ndw >> 16) & 0xffffu) | (1u << 16));    // tensor_dim0 hi | tensor_dim1=1
  g1[3] = (int)((ndw & 0xffffu) << 16);                   // tile_dim0 = 2048
  g1[4] = 1;                                              // tile_dim1 = 1, tile_dim2 = 0
  g1[5] = (int)ndw;                                       // tensor_dim0_stride
  g1[6] = 0;
  g1[7] = 0;
  i32x4 z4 = {0, 0, 0, 0};
#if defined(__clang_major__) && __clang_major__ >= 23
  i32x8 z8 = {0, 0, 0, 0, 0, 0, 0, 0};
  __builtin_amdgcn_tensor_load_to_lds(g0, g1, z4, z4, z8, 0);
#else
  __builtin_amdgcn_tensor_load_to_lds(g0, g1, z4, z4, 0);
#endif
}
#endif

// ---------------------------------------------------------------------------
// Fused: x_s = tanh(xm^T T xn + W[xm;xn] + b)  then  h = x_s @ W_gcn
// 256 threads = 8 waves; block covers 128 nodes; wave w owns rows w*16..+15.
// B chunks staged to LDS by the Tensor Data Mover, triple buffered.
// ---------------------------------------------------------------------------
__launch_bounds__(256, 2)
__global__ void fused_fusion_kernel(const float* __restrict__ xm,
                                    const float* __restrict__ xn,
                                    const float* __restrict__ bfuse,
                                    const __bf16* __restrict__ bp_main,
                                    const __bf16* __restrict__ bp_gcn,
                                    float* __restrict__ hout, int N) {
  __shared__ float  s_xm[128][DMKT + 1];       // +1: bank-conflict-free rows
  __shared__ float  s_xn[128][DMKT + 1];
  __shared__ __bf16 s_xs[128][DH + 2];         // 65-dword row stride
  __shared__ __align__(32) __bf16 s_b[3][CHUNK_BYTES / 2];  // B chunk buffers

  const int tid  = threadIdx.x;
  const int base = blockIdx.x * 128;
  const int wave = tid >> 5;
  const int lane = tid & 31;

#if HAVE_TDM
  unsigned sb_off[3];
#pragma unroll
  for (int i = 0; i < 3; ++i)
    sb_off[i] = (unsigned)(unsigned long long)(const void*)&s_b[i][0];
  if (wave == 0) {                              // prime the pipeline (depth 2)
    tdm_load_chunk(bp_main, sb_off[0]);
    tdm_load_chunk(bp_main + (size_t)DH * 32, sb_off[1]);
  }
#endif

  // Stage xm/xn tile (rows are contiguous 64-float runs); clamp tail rows.
  {
    int r  = tid >> 1;
    int ho = (tid & 1) * 32;
    int gr = base + r; if (gr >= N) gr = N - 1;
    const float4* pm = (const float4*)(xm + (size_t)gr * 64 + ho);
    const float4* pn = (const float4*)(xn + (size_t)gr * 64 + ho);
#pragma unroll
    for (int q = 0; q < 8; ++q) {
      float4 a = pm[q], b = pn[q];
      s_xm[r][ho + q * 4 + 0] = a.x; s_xm[r][ho + q * 4 + 1] = a.y;
      s_xm[r][ho + q * 4 + 2] = a.z; s_xm[r][ho + q * 4 + 3] = a.w;
      s_xn[r][ho + q * 4 + 0] = b.x; s_xn[r][ho + q * 4 + 1] = b.y;
      s_xn[r][ho + q * 4 + 2] = b.z; s_xn[r][ho + q * 4 + 3] = b.w;
    }
  }
  __syncthreads();

  const int  mrow  = lane & 15;        // A-row / B-col / C-col index
  const bool lo    = lane < 16;
  const int  base0 = lo ? 0 : 8;       // C M-offset (accumulator row base)
  const int  rowb  = wave * 16;

  const float* xmrow = &s_xm[rowb + mrow][0];
  const float* xnrow = &s_xn[rowb + mrow][0];

  v8f acc[8];
#pragma unroll
  for (int i = 0; i < 8; ++i) { v8f z = {}; acc[i] = z; }

  // Make chunk c visible in LDS and keep the TDM pipeline 2 deep.
  auto pipe = [&](int c, bool last) -> const __bf16* {
#if HAVE_TDM
    if (wave == 0) {
      if (last) __builtin_amdgcn_s_wait_tensorcnt(0);
      else      __builtin_amdgcn_s_wait_tensorcnt(1);  // in-order => chunk c done
    }
    __syncthreads();                                   // all waves see chunk c;
    if (wave == 0 && c + 2 < NCHUNK)                   // buf (c+2)%3 last read at c-1
      tdm_load_chunk(bp_main + (size_t)(c + 2) * DH * 32, sb_off[(c + 2) % 3]);
    return &s_b[c % 3][0];
#else
    __syncthreads();                                   // previous reads done
    ((ulonglong4*)&s_b[0][0])[tid] =
        ((const ulonglong4*)(bp_main + (size_t)c * DH * 32))[tid];
    __syncthreads();
    return &s_b[0][0];
#endif
  };

  // One K-chunk: build A frag (contiguous k = 16*half + i, as B), 8 WMMAs vs LDS B.
  auto do_chunk = [&](const float* srcrow, float scale, int off,
                      const __bf16* bchunk) {
    v16bf afrag;
#pragma unroll
    for (int i = 0; i < 8; ++i)
      afrag[i] = (__bf16)(scale * srcrow[off + (lo ? 0 : 16) + i]);
#pragma unroll
    for (int i = 0; i < 8; ++i)
      afrag[8 + i] = (__bf16)(scale * srcrow[off + (lo ? 0 : 16) + 8 + i]);
    v16bf bfr[8];
#pragma unroll
    for (int ht = 0; ht < 8; ++ht) {
      ulonglong4 braw = *(const ulonglong4*)(bchunk + (ht * 16 + mrow) * 32 + (lo ? 0 : 16));
      bfr[ht] = __builtin_bit_cast(v16bf, braw);
    }
#pragma unroll
    for (int ht = 0; ht < 8; ++ht)
      acc[ht] = __builtin_amdgcn_wmma_f32_16x16x32_bf16(
          false, afrag, false, bfr[ht], (short)0, acc[ht], false, false);
  };

  // 128 bilinear chunks: A = xm[m, c>>1] * xn[m, (c&1)*32 ..], branch-free.
  for (int c = 0; c < 128; ++c)
    do_chunk(xnrow, xmrow[c >> 1], (c & 1) << 5, pipe(c, false));
  // 4 linear-term chunks: A = [xm ; xn] columns, scale = 1.
  do_chunk(xmrow, 1.0f,  0, pipe(128, false));
  do_chunk(xmrow, 1.0f, 32, pipe(129, false));
  do_chunk(xnrow, 1.0f,  0, pipe(130, false));
  do_chunk(xnrow, 1.0f, 32, pipe(131, true));

  // ---- epilogue: bias + tanh -> LDS (bf16); rows owned by this wave only
#pragma unroll
  for (int ht = 0; ht < 8; ++ht) {
    int h = ht * 16 + mrow;
    float bb = bfuse[h];
#pragma unroll
    for (int r = 0; r < 8; ++r)
      s_xs[rowb + base0 + r][h] = (__bf16)tanhf(acc[ht][r] + bb);
  }

  // ---- GEMM2: g = x_s @ W_gcn  (same-wave LDS dependency; no barrier)
  v8f acc2[4];
#pragma unroll
  for (int i = 0; i < 4; ++i) { v8f z = {}; acc2[i] = z; }

#pragma unroll
  for (int c2 = 0; c2 < NCHUNK_GCN; ++c2) {
    const __bf16* xr = &s_xs[rowb + mrow][c2 * 32];
    v16bf afrag;
#pragma unroll
    for (int i = 0; i < 8; ++i) afrag[i] = xr[(lo ? 0 : 16) + i];
#pragma unroll
    for (int i = 0; i < 8; ++i) afrag[8 + i] = xr[(lo ? 0 : 16) + 8 + i];

    const __bf16* bp = bp_gcn + (size_t)c2 * DG * 32;
#pragma unroll
    for (int nt = 0; nt < 4; ++nt) {
      ulonglong4 braw = *(const ulonglong4*)(bp + (size_t)(nt * 16 + mrow) * 32 + (lo ? 0 : 16));
      v16bf bfrag = __builtin_bit_cast(v16bf, braw);
      acc2[nt] = __builtin_amdgcn_wmma_f32_16x16x32_bf16(
          false, afrag, false, bfrag, (short)0, acc2[nt], false, false);
    }
  }

  // ---- store h (pre-aggregation GCN features), fp32
#pragma unroll
  for (int nt = 0; nt < 4; ++nt) {
    int col = nt * 16 + mrow;
#pragma unroll
    for (int r = 0; r < 8; ++r) {
      int grow = base + rowb + base0 + r;
      if (grow < N) hout[(size_t)grow * 64 + col] = acc2[nt][r];
    }
  }
}

// ---------------------------------------------------------------------------
// Edge scatter: agg[dst] += h[src] * rsqrt(deg[src]) * rsqrt(deg[dst])
// 16 lanes per edge, float4 per lane; h/agg are L2-resident (25.6 MB each).
// ---------------------------------------------------------------------------
__global__ void scatter_kernel(const int* __restrict__ ei,
                               const float* __restrict__ deg,
                               const float* __restrict__ h,
                               float* __restrict__ agg, int E, int N) {
  int t = blockIdx.x * blockDim.x + threadIdx.x;
  int e = t >> 4;
  if (e >= E) return;
  int j = (t & 15) << 2;
  int s = ei[e], d = ei[E + e];
  if ((unsigned)s >= (unsigned)N || (unsigned)d >= (unsigned)N) return;
  float norm = __frsqrt_rn(deg[s]) * __frsqrt_rn(deg[d]);
  float4 hv = *(const float4*)(h + (size_t)s * 64 + j);
  float* ap = agg + (size_t)d * 64 + j;
  atomicAdd(ap + 0, hv.x * norm);
  atomicAdd(ap + 1, hv.y * norm);
  atomicAdd(ap + 2, hv.z * norm);
  atomicAdd(ap + 3, hv.w * norm);
}

// out[n] = b_head + sum_j relu(agg[n,j] + h[n,j]/deg[n] + b_gcn[j]) * W_head[j]
__global__ void head_kernel(const float* __restrict__ agg,
                            const float* __restrict__ h,
                            const float* __restrict__ deg,
                            const float* __restrict__ bgcn,
                            const float* __restrict__ Wh,
                            const float* __restrict__ bh,
                            float* __restrict__ out, int N) {
  int n = blockIdx.x * blockDim.x + threadIdx.x;
  if (n >= N) return;
  float inv = 1.0f / deg[n];           // self loop: dinv^2 = 1/deg
  float s = 0.0f;
#pragma unroll 4
  for (int j = 0; j < 64; ++j) {
    float g = agg[(size_t)n * 64 + j] + h[(size_t)n * 64 + j] * inv + bgcn[j];
    s += fmaxf(g, 0.0f) * Wh[j];
  }
  out[n] = s + bh[0];
}

static inline size_t align256(size_t x) { return (x + 255) & ~(size_t)255; }

extern "C" void kernel_launch(void* const* d_in, const int* in_sizes, int n_in,
                              void* d_out, int out_size, void* d_ws, size_t ws_size,
                              hipStream_t stream) {
  const float* xm = (const float*)d_in[0];
  const float* xn = (const float*)d_in[1];
  const int*   ei = (const int*)d_in[2];
  const float* Tf = (const float*)d_in[3];
  const float* Wf = (const float*)d_in[4];
  const float* bf = (const float*)d_in[5];
  const float* Wg = (const float*)d_in[6];
  const float* bg = (const float*)d_in[7];
  const float* Wh = (const float*)d_in[8];
  const float* bh = (const float*)d_in[9];
  const int N = in_sizes[0] / 64;
  const int E = in_sizes[2] / 2;

  char* ws = (char*)d_ws;
  size_t off = 0;
  __bf16* bp_main = (__bf16*)(ws + off); off += align256((size_t)NCHUNK * DH * 32 * 2);
  __bf16* bp_gcn  = (__bf16*)(ws + off); off += align256((size_t)NCHUNK_GCN * DG * 32 * 2);
  float* hbuf = (float*)(ws + off); off += align256((size_t)N * 64 * 4);
  float* deg  = (float*)(ws + off); off += align256((size_t)N * 4);
  float* agg  = (float*)(ws + off); off += align256((size_t)N * 64 * 4);
  (void)ws_size;

  float* out = (float*)d_out;
  auto cdiv = [](long long a, long long b) { return (unsigned)((a + b - 1) / b); };

  pack_main_kernel<<<cdiv((long long)NCHUNK * DH * 32, 256), 256, 0, stream>>>(Tf, Wf, bp_main);
  pack_gcn_kernel<<<cdiv((long long)NCHUNK_GCN * DG * 32, 256), 256, 0, stream>>>(Wg, bp_gcn);
  init_kernel<<<cdiv((long long)N * 64, 256), 256, 0, stream>>>(agg, deg, N);
  degree_kernel<<<cdiv(E, 256), 256, 0, stream>>>(ei, deg, E, N);
  fused_fusion_kernel<<<cdiv(N, 128), 256, 0, stream>>>(xm, xn, bf, bp_main, bp_gcn, hbuf, N);
  scatter_kernel<<<cdiv((long long)E * 16, 256), 256, 0, stream>>>(ei, deg, hbuf, agg, E, N);
  head_kernel<<<cdiv(N, 256), 256, 0, stream>>>(agg, hbuf, deg, bg, Wh, bh, out, N);
}
